// MambaBlock_2456721293314
// MI455X (gfx1250) — compile-verified, matrix operand order checked by reading, not run on hardware
//
#include <hip/hip_runtime.h>
#include <hip/hip_fp16.h>
#include <math.h>
#include <stdint.h>

// ---------------------------------------------------------------------------
// Bidirectional Mamba block for MI455X (gfx1250, wave32, WMMA 16x16x32 f16,
// TDM tensor_load_to_lds staging for GEMM B-operands)
// ---------------------------------------------------------------------------

typedef __attribute__((ext_vector_type(16))) _Float16 v16h;
typedef __attribute__((ext_vector_type(8)))  float    v8f;
typedef unsigned int u32x4 __attribute__((ext_vector_type(4)));
typedef int          i32x8 __attribute__((ext_vector_type(8)));
typedef int          i32x4 __attribute__((ext_vector_type(4)));

#define DIMC     384
#define DINNER   768
#define DSTATE   16
#define DCONV    4
#define DTRANK   24
#define SEQB     2
#define SEQL     1728        // 12*12*12
#define MROWS    (SEQB*SEQL) // 3456

#if defined(__HIP_DEVICE_COMPILE__) && __has_builtin(__builtin_amdgcn_tensor_load_to_lds) && __has_builtin(__builtin_amdgcn_s_wait_tensorcnt)
#define HAVE_TDM 1
#else
#define HAVE_TDM 0
#endif

// ---------------------------------------------------------------------------
// Pack/convert fp32 matrix -> fp16, zero-padding to (dR x dC)
// ---------------------------------------------------------------------------
__global__ __launch_bounds__(256)
void pack_f16_kernel(const float* __restrict__ src, _Float16* __restrict__ dst,
                     int sR, int sC, int dR, int dC) {
  int idx = blockIdx.x * blockDim.x + threadIdx.x;
  int total = dR * dC;
  if (idx >= total) return;
  int r = idx / dC, c = idx % dC;
  float v = (r < sR && c < sC) ? src[(size_t)r * sC + c] : 0.0f;
  dst[idx] = (_Float16)v;
}

// ---------------------------------------------------------------------------
// LayerNorm over channel dim. Input x in (B, C, L) layout, output (B*L, C) f16
// ---------------------------------------------------------------------------
__global__ __launch_bounds__(128)
void ln_kernel(const float* __restrict__ x, const float* __restrict__ g,
               const float* __restrict__ bb, _Float16* __restrict__ out,
               int Bn, int C, int L) {
  __shared__ float s1[128], s2[128];
  int row = blockIdx.x;                 // b*L + l
  int b = row / L, l = row % L;
  const float* xp = x + (size_t)b * C * L + l;   // stride L over channels
  float s = 0.f, ss = 0.f;
  for (int c = threadIdx.x; c < C; c += blockDim.x) {
    float v = xp[(size_t)c * L];
    s += v; ss += v * v;
  }
  s1[threadIdx.x] = s; s2[threadIdx.x] = ss;
  __syncthreads();
  for (int st = 64; st > 0; st >>= 1) {
    if ((int)threadIdx.x < st) {
      s1[threadIdx.x] += s1[threadIdx.x + st];
      s2[threadIdx.x] += s2[threadIdx.x + st];
    }
    __syncthreads();
  }
  float mu = s1[0] / (float)C;
  float var = s2[0] / (float)C - mu * mu;
  float rs = rsqrtf(var + 1e-5f);
  for (int c = threadIdx.x; c < C; c += blockDim.x) {
    float v = (xp[(size_t)c * L] - mu) * rs * g[c] + bb[c];
    out[(size_t)row * C + c] = (_Float16)v;
  }
}

// ---------------------------------------------------------------------------
// WMMA GEMM:  D = act(A(MxK,f16) * Bt(NxK,f16)^T + bias [+ Cprev])
//
// Block = 256 threads = 8 waves. Block tile 128(M) x 64(N): wave w owns rows
// [128*by + 16w, +16) and all 64 N columns -> 4 accumulators, 4 WMMA per
// A-fragment (4x operand reuse in registers).  The 64x64-half B chunk for the
// current K-slice is staged in LDS once per block (8x reuse across waves):
//   - TDM path: wave 0 issues tensor_load_to_lds with a 2-D descriptor
//     (data_size=2B, tile 64x64, dim0_stride=ldb), s_wait_tensorcnt 0, then
//     a workgroup barrier publishes the chunk.
//   - Fallback: cooperative global->LDS copy (one v16h = 32 bytes per thread).
// K tail (K%64, i.e. dt_proj K=24) uses guarded direct global loads.
// D layout per ISA: VGPR r, lanes 0-15 -> M=r, lanes 16-31 -> M=r+8.
// act: 0=none, 1=softplus, 2=gelu(exact)
// ---------------------------------------------------------------------------
__global__ __launch_bounds__(256)
void gemm_wmma_kernel(const _Float16* __restrict__ A, int lda,
                      const _Float16* __restrict__ Bt, int ldb,
                      float* __restrict__ Cf, _Float16* __restrict__ Ch, int ldc,
                      const float* __restrict__ bias,
                      int M, int N, int K, int act, int accum) {
  __shared__ _Float16 Bs[64][64];   // [n within tile][k within chunk]

  int lane = threadIdx.x & 31;
  int wave = threadIdx.x >> 5;
  int hs  = lane >> 4;       // half-wave select
  int lid = lane & 15;
  int tileM = blockIdx.y * 128;
  int tileN = blockIdx.x * 64;

  int aRow = tileM + wave * 16 + lid;
  const _Float16* aPtr = A + (size_t)aRow * lda + hs * 16;

  v8f acc[4] = {v8f{}, v8f{}, v8f{}, v8f{}};

  int kChunks = (K & ~63) / 64;
  for (int c = 0; c < kChunks; ++c) {
    int k0 = c * 64;
#if HAVE_TDM
    if (wave == 0) {
      // Build Tensor DMA descriptor (D#): 2-D tensor (dim1=N rows, dim0=K
      // halves), tile 64x64 starting at (tileN, k0), destination = Bs.
      unsigned lds_base = (unsigned)(uintptr_t)&Bs[0][0];
      unsigned long long ga =
          (unsigned long long)(uintptr_t)(Bt + (size_t)tileN * ldb + k0);
      u32x4 g0;
      g0.x = 1u;                                   // count=1, user mode
      g0.y = lds_base;                             // lds_addr (bytes)
      g0.z = (unsigned)ga;                         // global_addr[31:0]
      g0.w = (unsigned)((ga >> 32) & 0x1FFFFFFull) // global_addr[56:32]
             | (2u << 30);                         // type=2 ("image")
      i32x8 g1;
      g1[0] = 0x00010000;                          // data_size=1 (2 bytes)
      g1[1] = (int)(((unsigned)K & 0xFFFFu) << 16);        // tensor_dim0 lo16
      g1[2] = (int)(((unsigned)K >> 16) |
                    (((unsigned)N & 0xFFFFu) << 16));      // dim0 hi / dim1 lo
      g1[3] = (int)(((unsigned)N >> 16) | (64u << 16));    // dim1 hi, tile_dim0=64
      g1[4] = 64;                                  // tile_dim1=64, tile_dim2=0
      g1[5] = ldb;                                 // tensor_dim0_stride[31:0]
      g1[6] = 0;                                   // stride hi, dim1_stride lo
      g1[7] = 0;                                   // dim1_stride hi (unused 2D)
      i32x4 gz = {0, 0, 0, 0};
#if __clang_major__ >= 23
      i32x8 gz8 = {0, 0, 0, 0, 0, 0, 0, 0};
      __builtin_amdgcn_tensor_load_to_lds(g0, g1, gz, gz, gz8, 0);
#else
      __builtin_amdgcn_tensor_load_to_lds(g0, g1, gz, gz, 0);
#endif
      __builtin_amdgcn_s_wait_tensorcnt(0);
    }
#else
    {
      // cooperative fill: 256 threads x 16 halves = 64x64 chunk
      int rB  = threadIdx.x >> 2;
      int seg = (threadIdx.x & 3) * 16;
      *(v16h*)&Bs[rB][seg] =
          *(const v16h*)(Bt + (size_t)(tileN + rB) * ldb + k0 + seg);
    }
#endif
    __syncthreads();
#pragma unroll
    for (int ks = 0; ks < 2; ++ks) {
      v16h a = *(const v16h*)(aPtr + k0 + ks * 32);
      if (c + 1 < kChunks)
        __builtin_prefetch(aPtr + k0 + 64 + ks * 32, 0, 1); // global_prefetch_b8
#pragma unroll
      for (int t = 0; t < 4; ++t) {
        v16h b = *(const v16h*)&Bs[t * 16 + lid][ks * 32 + hs * 16];
        acc[t] = __builtin_amdgcn_wmma_f32_16x16x32_f16(
            false, a, false, b, (short)0, acc[t], false, false);
      }
    }
    __syncthreads();
  }

  // ---- guarded K tail (only dt_proj: K=24) ----
  for (int k0 = (K & ~63); k0 < K; k0 += 32) {
    v16h a;
    for (int i = 0; i < 16; ++i) {
      int k = k0 + hs * 16 + i;
      a[i] = (k < K) ? A[(size_t)aRow * lda + k] : (_Float16)0.f;
    }
#pragma unroll
    for (int t = 0; t < 4; ++t) {
      int col = tileN + t * 16 + lid;
      v16h b;
      for (int i = 0; i < 16; ++i) {
        int k = k0 + hs * 16 + i;
        b[i] = (k < K) ? Bt[(size_t)col * ldb + k] : (_Float16)0.f;
      }
      acc[t] = __builtin_amdgcn_wmma_f32_16x16x32_f16(
          false, a, false, b, (short)0, acc[t], false, false);
    }
  }

  // ---- epilogue ----
  int rowBase = tileM + wave * 16 + hs * 8;
#pragma unroll
  for (int t = 0; t < 4; ++t) {
    int col = tileN + t * 16 + lid;
#pragma unroll
    for (int r = 0; r < 8; ++r) {
      int rr = rowBase + r;
      float v = acc[t][r];
      if (accum) v += Cf[(size_t)rr * ldc + col];
      if (bias)  v += bias[col];
      if (act == 1) {                               // softplus
        v = (v > 20.f) ? v : log1pf(__expf(v));
      } else if (act == 2) {                        // exact gelu
        v = 0.5f * v * (1.0f + erff(v * 0.70710678118654752440f));
      }
      if (Cf) Cf[(size_t)rr * ldc + col] = v;
      if (Ch) Ch[(size_t)rr * ldc + col] = (_Float16)v;
    }
  }
}

// ---------------------------------------------------------------------------
// Depthwise causal conv (D_CONV=4) + SiLU.  dir=0: taps t-j (forward);
// dir=1: taps t+j (backward branch evaluated in original index order).
// ---------------------------------------------------------------------------
__global__ __launch_bounds__(256)
void conv_silu_kernel(const float* __restrict__ xz, const float* __restrict__ w,
                      const float* __restrict__ cb, _Float16* __restrict__ out,
                      int Bn, int L, int dir) {
  int idx = blockIdx.x * blockDim.x + threadIdx.x;
  int total = Bn * L * DINNER;
  if (idx >= total) return;
  int d = idx % DINNER;
  int t = (idx / DINNER) % L;
  int b = idx / (DINNER * L);
  float acc = cb[d];
#pragma unroll
  for (int j = 0; j < DCONV; ++j) {
    int tt = dir ? (t + j) : (t - j);
    if (tt >= 0 && tt < L)
      acc += xz[((size_t)(b * L + tt)) * 1536 + d] * w[d * DCONV + (DCONV - 1 - j)];
  }
  float sv = acc / (1.f + __expf(-acc));
  out[((size_t)(b * L + t)) * DINNER + d] = (_Float16)sv;
}

// ---------------------------------------------------------------------------
// Selective scan.  One wave handles 2 channels (16 lanes/channel, lane = state
// index n).  h <- exp(dt*A)*h + dt*B*xc ; y = (sum_n h*C) + xc*D, gated by
// silu(z).  dir=1 iterates l from L-1 down to 0 (flipped sequence).
// ---------------------------------------------------------------------------
__global__ __launch_bounds__(256)
void scan_kernel(const float* __restrict__ xz, const _Float16* __restrict__ xc,
                 const float* __restrict__ xdbl, const float* __restrict__ dt,
                 const float* __restrict__ A_log, const float* __restrict__ Dp,
                 _Float16* __restrict__ y, int Bn, int L, int dir) {
  int lane = threadIdx.x & 31;
  int gw = (blockIdx.x * blockDim.x + threadIdx.x) >> 5;
  int seq = gw * 2 + (lane >> 4);
  if (seq >= Bn * DINNER) return;
  int b = seq / DINNER, d = seq % DINNER;
  int n = lane & 15;

  float Aneg = -__expf(A_log[d * DSTATE + n]);
  float Dd = Dp[d];
  float h = 0.f;
  for (int i = 0; i < L; ++i) {
    int l = dir ? (L - 1 - i) : i;
    size_t row = (size_t)b * L + l;
    float dtv = dt[row * DINNER + d];
    float xcv = (float)xc[row * DINNER + d];
    float Bv = xdbl[row * 64 + DTRANK + n];            // B state input
    float Cv = xdbl[row * 64 + DTRANK + DSTATE + n];   // C readout
    h = __expf(dtv * Aneg) * h + (dtv * Bv) * xcv;
    float p = h * Cv;
    p += __shfl_xor(p, 1);
    p += __shfl_xor(p, 2);
    p += __shfl_xor(p, 4);
    p += __shfl_xor(p, 8);
    if (n == 0) {
      float zv = xz[row * 1536 + DINNER + d];
      float yo = (p + xcv * Dd) * (zv / (1.f + __expf(-zv)));
      y[row * DINNER + d] = (_Float16)yo;
    }
  }
}

// ---------------------------------------------------------------------------
// out[b,c,l] = xin[b,c,l] + seq[(b*L+l)*C + c]   (residual + re-transpose)
// ---------------------------------------------------------------------------
__global__ __launch_bounds__(256)
void resadd_kernel(const float* __restrict__ xin, const float* __restrict__ seq,
                   float* __restrict__ out, int Bn, int C, int L) {
  int idx = blockIdx.x * blockDim.x + threadIdx.x;
  int total = Bn * C * L;
  if (idx >= total) return;
  int l = idx % L;
  int c = (idx / L) % C;
  int b = idx / (L * C);
  out[idx] = xin[idx] + seq[((size_t)(b * L + l)) * C + c];
}

// ---------------------------------------------------------------------------
// Host-side orchestration
// ---------------------------------------------------------------------------
extern "C" void kernel_launch(void* const* d_in, const int* in_sizes, int n_in,
                              void* d_out, int out_size, void* d_ws, size_t ws_size,
                              hipStream_t stream) {
  (void)in_sizes; (void)n_in; (void)out_size; (void)ws_size;

  const float* x      = (const float*)d_in[0];
  const float* ln1_g  = (const float*)d_in[1];
  const float* ln1_b  = (const float*)d_in[2];
  const float* ln2_g  = (const float*)d_in[3];
  const float* ln2_b  = (const float*)d_in[4];
  const float* fc1_w  = (const float*)d_in[5];
  const float* fc1_b  = (const float*)d_in[6];
  const float* fc2_w  = (const float*)d_in[7];
  const float* fc2_b  = (const float*)d_in[8];
  const float* f_in_w   = (const float*)d_in[9];
  const float* f_conv_w = (const float*)d_in[10];
  const float* f_conv_b = (const float*)d_in[11];
  const float* f_xproj  = (const float*)d_in[12];
  const float* f_dt_w   = (const float*)d_in[13];
  const float* f_dt_b   = (const float*)d_in[14];
  const float* f_A_log  = (const float*)d_in[15];
  const float* f_D      = (const float*)d_in[16];
  const float* f_out_w  = (const float*)d_in[17];
  const float* b_in_w   = (const float*)d_in[18];
  const float* b_conv_w = (const float*)d_in[19];
  const float* b_conv_b = (const float*)d_in[20];
  const float* b_xproj  = (const float*)d_in[21];
  const float* b_dt_w   = (const float*)d_in[22];
  const float* b_dt_b   = (const float*)d_in[23];
  const float* b_A_log  = (const float*)d_in[24];
  const float* b_D      = (const float*)d_in[25];
  const float* b_out_w  = (const float*)d_in[26];

  const int Bn = SEQB, L = SEQL, C = DIMC, M = MROWS;

  // ---- workspace carve-out (256B-aligned) ----
  char* base = (char*)d_ws;
  size_t off = 0;
  auto alloc = [&](size_t bytes) -> void* {
    size_t a = (off + 255) & ~(size_t)255;
    off = a + bytes;
    return (void*)(base + a);
  };

  _Float16* xn1_h   = (_Float16*)alloc((size_t)M * C * 2);
  _Float16* f_in_h  = (_Float16*)alloc((size_t)1536 * 384 * 2);
  _Float16* b_in_h  = (_Float16*)alloc((size_t)1536 * 384 * 2);
  _Float16* f_xp_h  = (_Float16*)alloc((size_t)64 * 768 * 2);
  _Float16* b_xp_h  = (_Float16*)alloc((size_t)64 * 768 * 2);
  _Float16* f_dtw_h = (_Float16*)alloc((size_t)768 * 32 * 2);
  _Float16* b_dtw_h = (_Float16*)alloc((size_t)768 * 32 * 2);
  _Float16* f_ow_h  = (_Float16*)alloc((size_t)384 * 768 * 2);
  _Float16* b_ow_h  = (_Float16*)alloc((size_t)384 * 768 * 2);
  _Float16* fc1_h   = (_Float16*)alloc((size_t)1536 * 384 * 2);
  _Float16* fc2_h   = (_Float16*)alloc((size_t)384 * 1536 * 2);

  float*    xzF     = (float*)   alloc((size_t)M * 1536 * 4);
  float*    xzB     = (float*)   alloc((size_t)M * 1536 * 4);
  _Float16* xcF_h   = (_Float16*)alloc((size_t)M * 768 * 2);
  _Float16* xcB_h   = (_Float16*)alloc((size_t)M * 768 * 2);
  float*    xdF     = (float*)   alloc((size_t)M * 64 * 4);
  _Float16* xdF_h   = (_Float16*)alloc((size_t)M * 64 * 2);
  float*    xdB     = (float*)   alloc((size_t)M * 64 * 4);
  _Float16* xdB_h   = (_Float16*)alloc((size_t)M * 64 * 2);
  float*    dtF     = (float*)   alloc((size_t)M * 768 * 4);
  float*    dtB     = (float*)   alloc((size_t)M * 768 * 4);
  _Float16* yF_h    = (_Float16*)alloc((size_t)M * 768 * 2);
  _Float16* yB_h    = (_Float16*)alloc((size_t)M * 768 * 2);
  float*    out_seq = (float*)   alloc((size_t)M * 384 * 4);
  float*    x1      = (float*)   alloc((size_t)Bn * C * L * 4);
  _Float16* xn2_h   = (_Float16*)alloc((size_t)M * C * 2);
  _Float16* h_h     = (_Float16*)alloc((size_t)M * 1536 * 2);
  float*    mlp_f   = (float*)   alloc((size_t)M * 384 * 4);

  auto pack = [&](const float* src, _Float16* dst, int sR, int sC, int dR, int dC) {
    int total = dR * dC;
    pack_f16_kernel<<<(total + 255) / 256, 256, 0, stream>>>(src, dst, sR, sC, dR, dC);
  };
  auto gemm = [&](const _Float16* A, int lda, const _Float16* Bt, int ldb,
                  float* Cf, _Float16* Ch, int ldc, const float* bias,
                  int Mm, int Nn, int Kk, int act, int accum) {
    dim3 grid(Nn / 64, Mm / 128);
    gemm_wmma_kernel<<<grid, 256, 0, stream>>>(A, lda, Bt, ldb, Cf, Ch, ldc,
                                               bias, Mm, Nn, Kk, act, accum);
  };

  // ---- pack all weight matrices to f16 (zero-padded where needed) ----
  pack(f_in_w, f_in_h, 1536, 384, 1536, 384);
  pack(b_in_w, b_in_h, 1536, 384, 1536, 384);
  pack(f_xproj, f_xp_h, 56, 768, 64, 768);      // pad N 56 -> 64
  pack(b_xproj, b_xp_h, 56, 768, 64, 768);
  pack(f_dt_w, f_dtw_h, 768, 24, 768, 32);      // pad K 24 -> 32
  pack(b_dt_w, b_dtw_h, 768, 24, 768, 32);
  pack(f_out_w, f_ow_h, 384, 768, 384, 768);
  pack(b_out_w, b_ow_h, 384, 768, 384, 768);
  pack(fc1_w, fc1_h, 1536, 384, 1536, 384);
  pack(fc2_w, fc2_h, 384, 1536, 384, 1536);

  // ---- stage 1: LN1 -> (M x C) f16 ----
  ln_kernel<<<M, 128, 0, stream>>>(x, ln1_g, ln1_b, xn1_h, Bn, C, L);

  // ---- stage 2: in_proj GEMMs (flip commutes with row-wise matmul) ----
  gemm(xn1_h, 384, f_in_h, 384, xzF, nullptr, 1536, nullptr, M, 1536, 384, 0, 0);
  gemm(xn1_h, 384, b_in_h, 384, xzB, nullptr, 1536, nullptr, M, 1536, 384, 0, 0);

  // ---- stage 3: depthwise causal conv + SiLU ----
  {
    int total = Bn * L * DINNER;
    conv_silu_kernel<<<(total + 255) / 256, 256, 0, stream>>>(xzF, f_conv_w, f_conv_b, xcF_h, Bn, L, 0);
    conv_silu_kernel<<<(total + 255) / 256, 256, 0, stream>>>(xzB, b_conv_w, b_conv_b, xcB_h, Bn, L, 1);
  }

  // ---- stage 4: x_proj GEMMs -> (M x 64) [dt | B | C], f32 + f16 copies ----
  gemm(xcF_h, 768, f_xp_h, 768, xdF, xdF_h, 64, nullptr, M, 64, 768, 0, 0);
  gemm(xcB_h, 768, b_xp_h, 768, xdB, xdB_h, 64, nullptr, M, 64, 768, 0, 0);

  // ---- stage 5: dt_proj GEMMs (K=24 guarded tail) + softplus ----
  gemm(xdF_h, 64, f_dtw_h, 32, dtF, nullptr, 768, f_dt_b, M, 768, 24, 1, 0);
  gemm(xdB_h, 64, b_dtw_h, 32, dtB, nullptr, 768, b_dt_b, M, 768, 24, 1, 0);

  // ---- stage 6: selective scans (fwd l:0..L-1 ; bwd l:L-1..0) ----
  {
    int waves = (Bn * DINNER) / 2;            // 2 channels per wave
    int blocks = (waves * 32 + 255) / 256;    // = 96
    scan_kernel<<<blocks, 256, 0, stream>>>(xzF, xcF_h, xdF, dtF, f_A_log, f_D, yF_h, Bn, L, 0);
    scan_kernel<<<blocks, 256, 0, stream>>>(xzB, xcB_h, xdB, dtB, b_A_log, b_D, yB_h, Bn, L, 1);
  }

  // ---- stage 7: out_proj GEMMs, fwd writes then bwd accumulates ----
  gemm(yF_h, 768, f_ow_h, 768, out_seq, nullptr, 384, nullptr, M, 384, 768, 0, 0);
  gemm(yB_h, 768, b_ow_h, 768, out_seq, nullptr, 384, nullptr, M, 384, 768, 0, 1);

  // ---- stage 8: residual add (back to (B,C,L)) ----
  {
    int total = Bn * C * L;
    resadd_kernel<<<(total + 255) / 256, 256, 0, stream>>>(x, out_seq, x1, Bn, C, L);
  }

  // ---- stage 9: LN2 + MLP (fc1+GELU, fc2) ----
  ln_kernel<<<M, 128, 0, stream>>>(x1, ln2_g, ln2_b, xn2_h, Bn, C, L);
  gemm(xn2_h, 384, fc1_h, 384, nullptr, h_h, 1536, fc1_b, M, 1536, 384, 2, 0);
  gemm(h_h, 1536, fc2_h, 1536, mlp_f, nullptr, 384, fc2_b, M, 384, 1536, 0, 0);

  // ---- stage 10: final residual into d_out ((B,C,L) layout) ----
  {
    int total = Bn * C * L;
    resadd_kernel<<<(total + 255) / 256, 256, 0, stream>>>(x1, mlp_f, (float*)d_out, Bn, C, L);
  }
}
